// ImprovedAttention_59717225284287
// MI455X (gfx1250) — matrix operand order repaired and checked by reading, not compiled, not run on hardware
//
#include <hip/hip_runtime.h>
#include <cstdint>
#include <cstddef>

// ---------------------------------------------------------------------------
// Types for CDNA5 WMMA (wave32): v16bf A/B operands, v8f f32 accumulator.
// ---------------------------------------------------------------------------
typedef __attribute__((ext_vector_type(16))) __bf16 bf16x16;  // 32B, 8 VGPRs
typedef __attribute__((ext_vector_type(8)))  __bf16 bf16x8;   // 16B
typedef __attribute__((ext_vector_type(4)))  __bf16 bf16x4;   // 8B
typedef __attribute__((ext_vector_type(8)))  float  f32x8;    // 32B, 8 VGPRs

union FragU { bf16x16 f; bf16x8 h[2]; };

__device__ __forceinline__ f32x8 wmma_bf16(bf16x16 a, bf16x16 b, f32x8 c) {
  return __builtin_amdgcn_wmma_f32_16x16x32_bf16(
      /*neg_a=*/false, a, /*neg_b=*/false, b,
      /*c_mod=*/(short)0, c, /*reuse_a=*/false, /*reuse_b=*/false);
}

// A-operand (16x32, bf16): lane l -> row (l&15); elems 0..7 = K (l>>4)*8..,
// elems 8..15 = K 16+(l>>4)*8..  Two contiguous 16B loads per lane.
__device__ __forceinline__ bf16x16 load_a_frag(const __bf16* base, int ld, int lane) {
  const int l = lane & 15, hi = lane >> 4;
  const __bf16* p = base + l * ld + hi * 8;
  FragU u;
  u.h[0] = *(const bf16x8*)(p);
  u.h[1] = *(const bf16x8*)(p + 16);
  return u.f;
}

// B-operand (32x16, bf16): lane l -> col (l&15); elems 0..7 = K (l>>4)*8..,
// elems 8..15 = K 16+(l>>4)*8..  Two 16B loads per lane from an N-major (K contiguous) buffer.
__device__ __forceinline__ bf16x16 load_b_frag(const __bf16* base, int ld, int lane) {
  const int l = lane & 15, hi = lane >> 4;
  const __bf16* p = base + l * ld + hi * 8;
  FragU u;
  u.h[0] = *(const bf16x8*)(p);
  u.h[1] = *(const bf16x8*)(p + 16);
  return u.f;
}

// ---------------------------------------------------------------------------
// Problem dims
// ---------------------------------------------------------------------------
#define BATCH 2
#define SEQ   2048
#define HEADS 16
#define HDIM  128
#define EMBED (HEADS * HDIM)        // 2048
#define M_TOT (BATCH * SEQ)         // 4096
#define N_TOT (3 * EMBED)           // 6144

// ---------------------------------------------------------------------------
// Kernel 1: x (f32) -> bf16, same row-major layout [M_TOT][EMBED]
// ---------------------------------------------------------------------------
__global__ __launch_bounds__(256) void cvt_x_kernel(const float* __restrict__ x,
                                                    __bf16* __restrict__ xb, int n4) {
  int stride = gridDim.x * blockDim.x;
  for (int idx = blockIdx.x * blockDim.x + threadIdx.x; idx < n4; idx += stride) {
    float4 f = ((const float4*)x)[idx];
    bf16x4 o;
    o[0] = (__bf16)f.x; o[1] = (__bf16)f.y; o[2] = (__bf16)f.z; o[3] = (__bf16)f.w;
    *(bf16x4*)(xb + (size_t)idx * 4) = o;
  }
}

// ---------------------------------------------------------------------------
// Kernel 2: W [EMBED][N_TOT] f32 -> Wt [N_TOT][EMBED] bf16 (transpose + cast)
// ---------------------------------------------------------------------------
__global__ __launch_bounds__(256) void cvt_w_kernel(const float* __restrict__ W,
                                                    __bf16* __restrict__ Wt) {
  __shared__ float tile[32][33];
  const int n0 = blockIdx.x * 32;
  const int k0 = blockIdx.y * 32;
  const int tx = threadIdx.x & 31;
  const int ty = threadIdx.x >> 5;   // 0..7
#pragma unroll
  for (int i = 0; i < 32; i += 8)
    tile[ty + i][tx] = W[(size_t)(k0 + ty + i) * N_TOT + n0 + tx];
  __syncthreads();
#pragma unroll
  for (int i = 0; i < 32; i += 8)
    Wt[(size_t)(n0 + ty + i) * EMBED + k0 + tx] = (__bf16)tile[tx][ty + i];
}

// ---------------------------------------------------------------------------
// Kernel 3: QKV GEMM (bf16 WMMA, f32 acc) + bias + fused RoPE epilogue.
//   128x128 C tile per block, 8 waves (2x4), wave tile 64x32, BK=64.
//   Global->LDS via GLOBAL_LOAD_ASYNC_TO_LDS_B128, double-buffered: the
//   k-step i+1 tile streams in (ASYNCcnt) while step i's 16 WMMAs execute.
// Dynamic LDS map (no static shared, so dynamic base = LDS offset 0):
//   [0,      65536)  tileC f32 [128][128]   (epilogue only)
//   [65536, 131072)  2 x { A bf16[128][64] (16KB), B bf16[128][64] (16KB) }
// ---------------------------------------------------------------------------
__global__ __launch_bounds__(256) void qkv_gemm_rope_kernel(
    const __bf16* __restrict__ xb,   // [4096][2048]
    const __bf16* __restrict__ wt,   // [6144][2048]
    const float*  __restrict__ bias, // [6144]
    __bf16* __restrict__ Qb, __bf16* __restrict__ Kb, __bf16* __restrict__ Vb) {
  extern __shared__ char smem[];
  float* tileC = (float*)smem;

  const int tid  = threadIdx.x;
  const int lane = tid & 31, wave = tid >> 5;
  const int mw = wave >> 2, nw = wave & 3;            // waves: 2 (M) x 4 (N)
  const int m0 = blockIdx.y * 128;
  const int n0 = blockIdx.x * 128;

  // async tile fill: each thread owns one 64B span of a row (4 x b128)
  const int lrow = tid >> 1;            // 0..127
  const int lcol = (tid & 1) * 32;      // 0 / 32 (elements)

  auto issue_tiles = [&](uint32_t k0, uint32_t buf) {
    const uint32_t ba = 65536u + buf * 32768u + (uint32_t)(lrow * 64 + lcol) * 2u;
    const uint32_t bb = ba + 16384u;
    const uint32_t ga = ((uint32_t)(m0 + lrow) * (uint32_t)EMBED + k0 + lcol) * 2u;
    const uint32_t gb = ((uint32_t)(n0 + lrow) * (uint32_t)EMBED + k0 + lcol) * 2u;
    asm volatile(
        "global_load_async_to_lds_b128 %0, %1, %4\n"
        "global_load_async_to_lds_b128 %0, %1, %4 offset:16\n"
        "global_load_async_to_lds_b128 %0, %1, %4 offset:32\n"
        "global_load_async_to_lds_b128 %0, %1, %4 offset:48\n"
        "global_load_async_to_lds_b128 %2, %3, %5\n"
        "global_load_async_to_lds_b128 %2, %3, %5 offset:16\n"
        "global_load_async_to_lds_b128 %2, %3, %5 offset:32\n"
        "global_load_async_to_lds_b128 %2, %3, %5 offset:48\n"
        :
        : "v"(ba), "v"(ga), "v"(bb), "v"(gb), "s"(xb), "s"(wt)
        : "memory");
  };

  f32x8 acc[4][2];
#pragma unroll
  for (int i = 0; i < 4; ++i)
#pragma unroll
    for (int j = 0; j < 2; ++j)
#pragma unroll
      for (int r = 0; r < 8; ++r) acc[i][j][r] = 0.f;

  issue_tiles(0u, 0u);

  for (int k0 = 0; k0 < EMBED; k0 += 64) {
    const int buf = (k0 >> 6) & 1;
    asm volatile("s_wait_asynccnt 0" ::: "memory");  // this wave's fills done
    __syncthreads();                                 // all waves' fills visible
    if (k0 + 64 < EMBED) issue_tiles((uint32_t)(k0 + 64), (uint32_t)(buf ^ 1));

    const __bf16* sA = (const __bf16*)(smem + 65536 + buf * 32768);
    const __bf16* sB = sA + 8192;   // +16KB
#pragma unroll
    for (int dk = 0; dk < 64; dk += 32) {
      bf16x16 a[4];
#pragma unroll
      for (int i = 0; i < 4; ++i)
        a[i] = load_a_frag(sA + (mw * 64 + i * 16) * 64 + dk, 64, lane);
#pragma unroll
      for (int j = 0; j < 2; ++j) {
        bf16x16 b = load_b_frag(sB + (nw * 32 + j * 16) * 64 + dk, 64, lane);
#pragma unroll
        for (int i = 0; i < 4; ++i)
          acc[i][j] = wmma_bf16(a[i], b, acc[i][j]);
      }
    }
  }

  // Epilogue: stage C tile in LDS, then RoPE + cast + scatter to Q/K/V.
  {
    const int l = lane & 15, hi = lane >> 4;
#pragma unroll
    for (int i = 0; i < 4; ++i)
#pragma unroll
      for (int j = 0; j < 2; ++j) {
        const int n = nw * 32 + j * 16 + l;
#pragma unroll
        for (int r = 0; r < 8; ++r) {
          const int m = mw * 64 + i * 16 + r + 8 * hi;
          tileC[m * 128 + n] = acc[i][j][r];
        }
      }
  }
  __syncthreads();

  const int c = n0 >> 11;             // 0=q, 1=k, 2=v
  const int h = (n0 & 2047) >> 7;     // head index (constant per block)
  __bf16* dst = (c == 0) ? Qb : (c == 1) ? Kb : Vb;

  for (int idx = tid; idx < 128 * 128; idx += 256) {
    const int m = idx >> 7, d = idx & 127;
    const int row = m0 + m;
    const int bb = row >> 11;         // batch
    const int s  = row & 2047;        // seq position
    float v = tileC[m * 128 + d] + bias[n0 + d];
    float outv;
    if (c < 2) {
      const float partner = tileC[m * 128 + (d ^ 64)] + bias[n0 + (d ^ 64)];
      const float ang = (float)(s * HDIM + d);
      const float co = __cosf(ang), si = __sinf(ang);
      const float rot = (d < 64) ? -partner : partner;
      outv = v * co + rot * si;
    } else {
      outv = v;
    }
    dst[(((size_t)bb * HEADS + h) * SEQ + s) * HDIM + d] = (__bf16)outv;
  }
}

// ---------------------------------------------------------------------------
// Kernel 4: flash attention. Grid (SEQ/128, B*H). 8 waves/block; each wave
// owns 16 query rows. KV blocks of 64. Q/K fragments read straight from
// global (layouts match WMMA striping). V block streams into an LDS staging
// buffer via GLOBAL_LOAD_ASYNC_TO_LDS_B128 *before* the QK^T WMMAs, then is
// transposed in-LDS so P*V B-fragments are contiguous 16B reads.
// Dynamic LDS map:
//   [0,     16384)  V stage bf16 [64 kv][128 d]
//   [16384, 32768)  Vt       bf16 [128 d][64 kv]
//   [32768, 49152)  P stage  bf16 8 waves x [16][64]
// ---------------------------------------------------------------------------
__global__ __launch_bounds__(256) void flash_attn_kernel(
    const __bf16* __restrict__ Qb, const __bf16* __restrict__ Kb,
    const __bf16* __restrict__ Vb, float* __restrict__ out) {
  extern __shared__ char smem[];
  __bf16* sVstage = (__bf16*)smem;
  __bf16* sVt     = (__bf16*)(smem + 16384);
  __bf16* sP      = (__bf16*)(smem + 32768);

  const int tid  = threadIdx.x;
  const int lane = tid & 31, wave = tid >> 5;
  const int l = lane & 15, hi = lane >> 4;
  const int bh = blockIdx.y;              // 0..31
  const int b  = bh >> 4, h = bh & 15;
  const int q0 = blockIdx.x * 128 + wave * 16;

  const __bf16* qp = Qb + (size_t)bh * SEQ * HDIM;
  const __bf16* kp = Kb + (size_t)bh * SEQ * HDIM;
  const __bf16* vp = Vb + (size_t)bh * SEQ * HDIM;
  __bf16* pst = sP + wave * 16 * 64;

  // Q A-fragments, loaded once (d-contraction split into 4 k-steps of 32)
  bf16x16 aq[4];
#pragma unroll
  for (int dk = 0; dk < 4; ++dk)
    aq[dk] = load_a_frag(qp + (size_t)q0 * HDIM + dk * 32, HDIM, lane);

  f32x8 o[8];
  float mrow[8], lrowv[8];
#pragma unroll
  for (int t = 0; t < 8; ++t)
#pragma unroll
    for (int r = 0; r < 8; ++r) o[t][r] = 0.f;
#pragma unroll
  for (int r = 0; r < 8; ++r) { mrow[r] = -1e30f; lrowv[r] = 0.f; }

  const float scale = 0.08838834764831845f;  // 1/sqrt(128)

  for (int kv0 = 0; kv0 < SEQ; kv0 += 64) {
    __syncthreads();  // prev iteration's sVt / stage readers are done

    // Kick off async V block copy (64x128 bf16 = 16KB, 4 x b128 per thread);
    // stage layout matches linear global order, so lds_off == local goff.
    {
      const uint32_t g0 = (uint32_t)kv0 * (uint32_t)HDIM * 2u;
#pragma unroll
      for (int ch = 0; ch < 4; ++ch) {
        const uint32_t linear = (uint32_t)tid + (uint32_t)ch * 256u;
        const uint32_t ldso = linear * 16u;
        const uint32_t goff = g0 + linear * 16u;
        asm volatile("global_load_async_to_lds_b128 %0, %1, %2"
                     :
                     : "v"(ldso), "v"(goff), "s"(vp)
                     : "memory");
      }
    }

    // S = Q * K^T for 16 q-rows x 64 kv-cols (B frags straight from global K)
    f32x8 s[4];
#pragma unroll
    for (int j = 0; j < 4; ++j)
#pragma unroll
      for (int r = 0; r < 8; ++r) s[j][r] = 0.f;
#pragma unroll
    for (int dk = 0; dk < 4; ++dk) {
#pragma unroll
      for (int j = 0; j < 4; ++j) {
        bf16x16 kb = load_b_frag(kp + (size_t)(kv0 + j * 16) * HDIM + dk * 32,
                                 HDIM, lane);
        s[j] = wmma_bf16(aq[dk], kb, s[j]);
      }
    }

    asm volatile("s_wait_asynccnt 0" ::: "memory");  // V stage (this wave) done
    __syncthreads();                                 // all waves' stage visible

    // In-LDS transpose: stage[kv][d] -> sVt[d][kv]
#pragma unroll
    for (int ch = 0; ch < 4; ++ch) {
      const int linear = tid + ch * 256;       // 0..1023
      const int vrow = linear >> 4;            // 0..63  (kv)
      const int vcol = (linear & 15) * 8;      // 0..120 (d)
      bf16x8 d8 = *(const bf16x8*)(sVstage + vrow * 128 + vcol);
#pragma unroll
      for (int e = 0; e < 8; ++e) sVt[(vcol + e) * 64 + vrow] = d8[e];
    }
    __syncthreads();  // sVt ready for all waves

    // Online softmax over rows (r + 8*hi); reductions across 16-lane groups
    float p[4][8], alpha[8];
#pragma unroll
    for (int r = 0; r < 8; ++r) {
      float mx = -1e30f;
#pragma unroll
      for (int j = 0; j < 4; ++j) mx = fmaxf(mx, s[j][r]);
      mx = fmaxf(mx, __shfl_xor(mx, 1, 32));
      mx = fmaxf(mx, __shfl_xor(mx, 2, 32));
      mx = fmaxf(mx, __shfl_xor(mx, 4, 32));
      mx = fmaxf(mx, __shfl_xor(mx, 8, 32));
      mx *= scale;
      const float mnew = fmaxf(mrow[r], mx);
      alpha[r] = __expf(mrow[r] - mnew);
      float rs = 0.f;
#pragma unroll
      for (int j = 0; j < 4; ++j) {
        const float pv = __expf(s[j][r] * scale - mnew);
        p[j][r] = pv;
        rs += pv;
      }
      rs += __shfl_xor(rs, 1, 32);
      rs += __shfl_xor(rs, 2, 32);
      rs += __shfl_xor(rs, 4, 32);
      rs += __shfl_xor(rs, 8, 32);
      lrowv[r] = lrowv[r] * alpha[r] + rs;
      mrow[r] = mnew;
    }
#pragma unroll
    for (int t = 0; t < 8; ++t)
#pragma unroll
      for (int r = 0; r < 8; ++r) o[t][r] *= alpha[r];

    // Restage P (C layout) -> A layout via per-wave LDS scratch
#pragma unroll
    for (int j = 0; j < 4; ++j)
#pragma unroll
      for (int r = 0; r < 8; ++r)
        pst[(r + 8 * hi) * 64 + j * 16 + l] = (__bf16)p[j][r];
    asm volatile("s_wait_dscnt 0" ::: "memory");  // same-wave DS write->read
    bf16x16 ap0 = load_a_frag(pst + 0,  64, lane);
    bf16x16 ap1 = load_a_frag(pst + 32, 64, lane);

    // O += P * V  (B frags are contiguous rows of transposed V in LDS)
#pragma unroll
    for (int t = 0; t < 8; ++t) {
      bf16x16 b0 = load_b_frag(sVt + t * 16 * 64 + 0,  64, lane);
      o[t] = wmma_bf16(ap0, b0, o[t]);
      bf16x16 b1 = load_b_frag(sVt + t * 16 * 64 + 32, 64, lane);
      o[t] = wmma_bf16(ap1, b1, o[t]);
    }
  }

  // Normalize and store: out[b][s][h][d] (f32)
#pragma unroll
  for (int t = 0; t < 8; ++t) {
    const int d = t * 16 + l;
#pragma unroll
    for (int r = 0; r < 8; ++r) {
      const int sq = q0 + r + 8 * hi;
      const float val = o[t][r] / lrowv[r];
      out[(((size_t)b * SEQ + sq) * HEADS + h) * HDIM + d] = val;
    }
  }
}

// ---------------------------------------------------------------------------
// Host launcher
// ---------------------------------------------------------------------------
extern "C" void kernel_launch(void* const* d_in, const int* in_sizes, int n_in,
                              void* d_out, int out_size, void* d_ws, size_t ws_size,
                              hipStream_t stream) {
  (void)in_sizes; (void)n_in; (void)out_size; (void)ws_size;
  const float* x    = (const float*)d_in[0];
  const float* W    = (const float*)d_in[1];
  const float* bias = (const float*)d_in[2];
  float* out = (float*)d_out;

  char* ws = (char*)d_ws;
  __bf16* xb = (__bf16*)ws;                                 // 16,777,216 B
  __bf16* wt = (__bf16*)(ws + (size_t)16777216);            // 25,165,824 B
  __bf16* Qb = (__bf16*)(ws + (size_t)16777216 + 25165824); // 16,777,216 B each
  __bf16* Kb = Qb + (size_t)BATCH * HEADS * SEQ * HDIM;
  __bf16* Vb = Kb + (size_t)BATCH * HEADS * SEQ * HDIM;

  (void)hipFuncSetAttribute((const void*)qkv_gemm_rope_kernel,
                            hipFuncAttributeMaxDynamicSharedMemorySize, 131072);
  (void)hipFuncSetAttribute((const void*)flash_attn_kernel,
                            hipFuncAttributeMaxDynamicSharedMemorySize, 49152);

  // 1. cast x -> bf16
  const int n4 = (M_TOT * EMBED) / 4;
  cvt_x_kernel<<<2048, 256, 0, stream>>>(x, xb, n4);
  // 2. transpose+cast W -> Wt[N][K] bf16
  cvt_w_kernel<<<dim3(N_TOT / 32, EMBED / 32), 256, 0, stream>>>(W, wt);
  // 3. QKV GEMM + RoPE, writes Q/K/V bf16 [B][H][S][D]
  qkv_gemm_rope_kernel<<<dim3(N_TOT / 128, M_TOT / 128), 256, 131072, stream>>>(
      xb, wt, bias, Qb, Kb, Vb);
  // 4. flash attention -> out f32 [B][S][H][D]
  flash_attn_kernel<<<dim3(SEQ / 128, BATCH * HEADS), 256, 49152, stream>>>(
      Qb, Kb, Vb, out);
}
